// ProteinFuncAttention_55980603736416
// MI455X (gfx1250) — matrix operand order repaired and checked by reading, not compiled, not run on hardware
//
#include <hip/hip_runtime.h>
#include <hip/hip_bf16.h>
#include <math.h>

// ---------------------------------------------------------------------------
// Model constants (reference: B=4, S=2048, D=1024, H=16, L=8, DH=64)
// ---------------------------------------------------------------------------
#define BB   4
#define SS   2048
#define DD   1024
#define HH   16
#define DHH  64
#define LL   8
#define MM   (BB * SS)          // 8192 token rows
#define EPS  1e-5f

typedef __attribute__((ext_vector_type(16))) __bf16 bf16x16;
typedef __attribute__((ext_vector_type(8)))  float  v8f;

// LDS row stride in bf16 elements: 32 data + 2 pad = 68B = 17 banks (odd -> no
// bank conflicts across the 16 lanes of a fragment read; pairs stay 4B aligned)
#define LPAD 34

__device__ __forceinline__ unsigned short f2bf(float f) {
    union { float f; unsigned u; } c; c.f = f;
    unsigned u = c.u;
    unsigned r = u + 0x7FFFu + ((u >> 16) & 1u);   // round-to-nearest-even
    return (unsigned short)(r >> 16);
}

// ---------------------------------------------------------------------------
// Tiled bf16 WMMA GEMM:  C[M,N] = act( A[M,K] @ W[K,N] + bias )
// block = 256 threads (8 waves), block tile 128x128, K-step 32.
// wave grid 2(M) x 4(N): each wave computes 64x32 = 4x2 WMMA tiles.
// 2-stage pipeline: double-buffered LDS, global float4 prefetch for tile k+1
// issued before the WMMAs of tile k; one barrier per iteration.
// Requires M%128==0, N%128==0, K%32==0 (true for all large GEMMs here).
// ACT: 0 = none, 1 = tanh
// ---------------------------------------------------------------------------
template <int ACT>
__global__ __launch_bounds__(256) void gemm_bf16_wmma(
    const float* __restrict__ A, const float* __restrict__ W,
    const float* __restrict__ bias, float* __restrict__ C,
    int M, int N, int K)
{
    __shared__ unsigned short lsA[2][128 * LPAD];   // [m][k]  bf16
    __shared__ unsigned short lsB[2][128 * LPAD];   // [n][k]  bf16 (transposed)

    const int tid    = threadIdx.x;
    const int lane   = tid & 31;
    const int wave   = tid >> 5;
    const int waveM  = wave >> 2;       // 0..1
    const int waveN  = wave & 3;        // 0..3
    const int halfId = lane >> 4;       // 0/1
    const int l16    = lane & 15;

    const int bm = blockIdx.y * 128;
    const int bn = blockIdx.x * 128;

    // per-thread staging coordinates (it = 0..3, 1024 float4 groups per tile)
    int rowA[4], c4A[4], krB[4], n4B[4];
#pragma unroll
    for (int it = 0; it < 4; ++it) {
        const int g = it * 256 + tid;
        rowA[it] = g >> 3;            // 0..127
        c4A[it]  = (g & 7) << 2;      // 0..28 step 4
        krB[it]  = g >> 5;            // 0..31
        n4B[it]  = (g & 31) << 2;     // 0..124 step 4
    }

    v8f acc[4][2];
#pragma unroll
    for (int i = 0; i < 4; ++i)
#pragma unroll
        for (int j = 0; j < 2; ++j)
            acc[i][j] = (v8f){0.f,0.f,0.f,0.f,0.f,0.f,0.f,0.f};

    float4 pfA[4], pfB[4];

    // global fetch of K-tile at k0 into registers (b128 loads, coalesced)
    auto loadG = [&](int k0) {
#pragma unroll
        for (int it = 0; it < 4; ++it)
            pfA[it] = *(const float4*)(A + (size_t)(bm + rowA[it]) * K + k0 + c4A[it]);
#pragma unroll
        for (int it = 0; it < 4; ++it)
            pfB[it] = *(const float4*)(W + (size_t)(k0 + krB[it]) * N + bn + n4B[it]);
    };
    // convert fp32 -> bf16 and stage into LDS buffer `buf`
    auto stageL = [&](int buf) {
#pragma unroll
        for (int it = 0; it < 4; ++it) {
            unsigned short* p = &lsA[buf][rowA[it] * LPAD + c4A[it]];
            const float4 f = pfA[it];
            *(unsigned*)(p)     = (unsigned)f2bf(f.x) | ((unsigned)f2bf(f.y) << 16);
            *(unsigned*)(p + 2) = (unsigned)f2bf(f.z) | ((unsigned)f2bf(f.w) << 16);
        }
#pragma unroll
        for (int it = 0; it < 4; ++it) {
            const float4 f = pfB[it];
            unsigned short* q = &lsB[buf][n4B[it] * LPAD + krB[it]];
            q[0 * LPAD] = f2bf(f.x);
            q[1 * LPAD] = f2bf(f.y);
            q[2 * LPAD] = f2bf(f.z);
            q[3 * LPAD] = f2bf(f.w);
        }
    };

    const int kTiles = K >> 5;

    loadG(0);
    stageL(0);
    __syncthreads();

    for (int kk = 0; kk < kTiles; ++kk) {
        const int cur = kk & 1;

        // ---- issue global prefetch for tile kk+1 (in flight during WMMAs)
        if (kk + 1 < kTiles) loadG((kk + 1) << 5);

        // ---- load fragments (per-lane layout per CDNA5 ISA 7.12.2) ------
        union Frag { bf16x16 v; unsigned u[8]; };
        Frag afr[4], bfr[2];
#pragma unroll
        for (int mt = 0; mt < 4; ++mt) {
            const int row = waveM * 64 + mt * 16 + l16;
            const unsigned short* base = &lsA[cur][row * LPAD];
#pragma unroll
            for (int v = 0; v < 8; ++v) {
                // A 16x32: VGPR v holds K = {k, k+1}; k = (v&4)*4 + half*8 + (v&3)*2
                int k = ((v & 4) << 2) + (halfId << 3) + ((v & 3) << 1);
                afr[mt].u[v] = *(const unsigned*)(base + k);
            }
        }
#pragma unroll
        for (int nt = 0; nt < 2; ++nt) {
            const int n = waveN * 32 + nt * 16 + l16;
            const unsigned short* base = &lsB[cur][n * LPAD];
#pragma unroll
            for (int v = 0; v < 8; ++v) {
                // B 32x16: VGPR v holds K = {k, k+1}; k = (v&4)*4 + half*8 + (v&3)*2
                int k = ((v & 4) << 2) + (halfId << 3) + ((v & 3) << 1);
                bfr[nt].u[v] = *(const unsigned*)(base + k);
            }
        }
        // ---- 8 WMMAs ----------------------------------------------------
#pragma unroll
        for (int mt = 0; mt < 4; ++mt)
#pragma unroll
            for (int nt = 0; nt < 2; ++nt)
                acc[mt][nt] = __builtin_amdgcn_wmma_f32_16x16x32_bf16(
                    false, afr[mt].v, false, bfr[nt].v,
                    (short)0, acc[mt][nt], false, false);

        // ---- stage tile kk+1 into the other buffer ----------------------
        // (safe without a pre-barrier: its last readers synced at the end of
        //  iteration kk-1)
        if (kk + 1 < kTiles) stageL((kk + 1) & 1);
        __syncthreads();
    }

    // ---- epilogue: bias + activation + fp32 store -----------------------
#pragma unroll
    for (int mt = 0; mt < 4; ++mt) {
#pragma unroll
        for (int nt = 0; nt < 2; ++nt) {
            const int gn = bn + waveN * 32 + nt * 16 + l16;
            const float bv = bias ? bias[gn] : 0.0f;
#pragma unroll
            for (int v = 0; v < 8; ++v) {
                const int gm = bm + waveM * 64 + mt * 16 + halfId * 8 + v;
                float val = acc[mt][nt][v] + bv;
                if (ACT == 1) val = tanhf(val);
                C[(size_t)gm * N + gn] = val;
            }
        }
    }
}

// ---------------------------------------------------------------------------
// Tiny GEMM for the M=4 timestep path (WMMA would waste 97% of the tile).
// ACT: 0 = none, 2 = silu
// ---------------------------------------------------------------------------
template <int ACT>
__global__ void gemm_small(const float* __restrict__ A, const float* __restrict__ W,
                           const float* __restrict__ bias, float* __restrict__ C,
                           int N, int K)
{
    int n = blockIdx.x * blockDim.x + threadIdx.x;
    int m = blockIdx.y;
    if (n >= N) return;
    float s = bias ? bias[n] : 0.0f;
    for (int k = 0; k < K; ++k) s += A[m * K + k] * W[(size_t)k * N + n];
    if (ACT == 2) s = s / (1.0f + __expf(-s));     // silu
    C[(size_t)m * N + n] = s;
}

// ---------------------------------------------------------------------------
// Sinusoidal timestep embedding: out[b, i] = cos/sin(t[b] * freq)
// ---------------------------------------------------------------------------
__global__ void pos_timestep_kernel(const int* __restrict__ t, float* __restrict__ out)
{
    int i = blockIdx.x * blockDim.x + threadIdx.x;   // 0..1023
    int b = blockIdx.y;
    const int half = DD / 2;
    float tv = (float)t[b];
    int j = (i < half) ? i : (i - half);
    float freq = __expf(-logf(10000.0f) * (float)j / (float)half);
    float a = tv * freq;
    out[b * DD + i] = (i < half) ? cosf(a) : sinf(a);
}

// ---------------------------------------------------------------------------
// Fused residual + broadcast + LayerNorm over D=1024. One block per row.
// out = LN(x [+ res] [+ bcast[row/bdiv]]) * g + b      (in-place safe)
// ---------------------------------------------------------------------------
__global__ __launch_bounds__(256) void ln_kernel(
    const float* __restrict__ x, const float* __restrict__ res,
    const float* __restrict__ bcast, int bdiv,
    const float* __restrict__ g, const float* __restrict__ bb,
    float* __restrict__ out)
{
    const int row = blockIdx.x;
    const int tid = threadIdx.x;
    __shared__ float red[256];

    const size_t base   = (size_t)row * DD;
    const size_t bcbase = bcast ? (size_t)(row / bdiv) * DD : 0;

    float v[4];
    float s = 0.f;
#pragma unroll
    for (int i = 0; i < 4; ++i) {
        int c = tid + i * 256;
        float val = x[base + c];
        if (res)   val += res[base + c];
        if (bcast) val += bcast[bcbase + c];
        v[i] = val;
        s += val;
    }
    red[tid] = s; __syncthreads();
    for (int off = 128; off > 0; off >>= 1) {
        if (tid < off) red[tid] += red[tid + off];
        __syncthreads();
    }
    const float mean = red[0] * (1.0f / DD);
    __syncthreads();

    float q = 0.f;
#pragma unroll
    for (int i = 0; i < 4; ++i) { float d = v[i] - mean; q += d * d; }
    red[tid] = q; __syncthreads();
    for (int off = 128; off > 0; off >>= 1) {
        if (tid < off) red[tid] += red[tid + off];
        __syncthreads();
    }
    const float rstd = rsqrtf(red[0] * (1.0f / DD) + EPS);

#pragma unroll
    for (int i = 0; i < 4; ++i) {
        int c = tid + i * 256;
        out[base + c] = (v[i] - mean) * rstd * g[c] + bb[c];
    }
}

// ---------------------------------------------------------------------------
// Per-position head-head attention + output scramble.
// For each (b,s): q,k,v = qkv row [16,64] slices; scores = qk^T/8 [16,16];
// softmax rows; o = a@v [16,64]; scatter per reference transpose/view:
//   out[b, h*S*64 + s*64 + dh]
// ---------------------------------------------------------------------------
__global__ __launch_bounds__(256) void attn_kernel(
    const float* __restrict__ qkv, float* __restrict__ out)
{
    const int pos = blockIdx.x;            // b*S + s
    const int b = pos / SS, s = pos % SS;
    const int tid = threadIdx.x;

    __shared__ float sq[HH * DHH], sk[HH * DHH], sv[HH * DHH], sa[HH * HH];

    const size_t base = (size_t)pos * (3 * DD);
    for (int i = tid; i < DD; i += 256) {
        sq[i] = qkv[base + i];
        sk[i] = qkv[base + DD + i];
        sv[i] = qkv[base + 2 * DD + i];
    }
    __syncthreads();

    // scores: 256 threads = 16h x 16g
    const int h = tid >> 4, gi = tid & 15;
    float sc = 0.f;
#pragma unroll 8
    for (int d = 0; d < DHH; ++d) sc += sq[h * DHH + d] * sk[gi * DHH + d];
    sc *= 0.125f;                                   // 1/sqrt(64)

    // softmax across the 16-thread group (width-16 shuffles within wave32)
    float mx = sc;
    for (int o = 8; o > 0; o >>= 1) mx = fmaxf(mx, __shfl_xor(mx, o, 16));
    float e = __expf(sc - mx);
    float sum = e;
    for (int o = 8; o > 0; o >>= 1) sum += __shfl_xor(sum, o, 16);
    sa[h * HH + gi] = e / sum;
    __syncthreads();

    // o = a @ v, 4 outputs per thread; write through the scramble permutation
#pragma unroll
    for (int e2 = 0; e2 < 4; ++e2) {
        int idx = tid * 4 + e2;                     // h2*64 + dh
        int h2 = idx >> 6, dh = idx & 63;
        float o = 0.f;
#pragma unroll
        for (int g2 = 0; g2 < HH; ++g2) o += sa[h2 * HH + g2] * sv[g2 * DHH + dh];
        size_t flat = (size_t)h2 * ((size_t)SS * DHH) + (size_t)s * DHH + dh;
        out[(size_t)b * ((size_t)SS * DD) + flat] = o;
    }
}

// ---------------------------------------------------------------------------
// Host orchestration
// ---------------------------------------------------------------------------
// Flattened input order (setup_inputs dict traversal):
//  0:x  1:timestep
//  2:t1.w 3:t1.b 4:t2.w 5:t2.b 6:lnA.g 7:lnA.b
//  8:in1.w 9:in1.b 10:in2.w 11:in2.b 12:lnB.g 13:lnB.b
//  14 + 11*l + {0:wqkv.w 1:wqkv.b 2:wout.w 3:ln1.g 4:ln1.b 5:ln2.g 6:ln2.b
//               7:ff1.w 8:ff1.b 9:ff2.w 10:ff2.b}
//  102:out1.w 103:out1.b 104:out2.w 105:out2.b 106:lnC.g 107:lnC.b

extern "C" void kernel_launch(void* const* d_in, const int* in_sizes, int n_in,
                              void* d_out, int out_size, void* d_ws, size_t ws_size,
                              hipStream_t stream)
{
    (void)in_sizes; (void)n_in; (void)out_size; (void)ws_size;
    auto F = [&](int i) { return (const float*)d_in[i]; };

    const float* x     = F(0);
    const int*   tstep = (const int*)d_in[1];

    // workspace layout (floats)
    float* qkv = (float*)d_ws;                 // 8192*3072
    float* xe  = qkv + (size_t)MM * 3 * DD;    // 8192*1024
    float* ta  = xe  + (size_t)MM * DD;        // 8192*1024
    float* tb  = ta  + (size_t)MM * DD;        // 8192*1024
    float* sm0 = tb  + (size_t)MM * DD;        // pos   [4,1024]
    float* sm1 = sm0 + BB * DD;                // h1    [4,4096]
    float* sm2 = sm1 + BB * 4 * DD;            // te0   [4,1024]
    float* te  = sm2 + BB * DD;                // te    [4,1024]

    const dim3 blk(256);
    const dim3 gD (DD / 128,      MM / 128);   // N=1024 GEMMs
    const dim3 g3D(3 * DD / 128,  MM / 128);   // N=3072 GEMM

    // ---- timestep embedding path (tiny) --------------------------------
    pos_timestep_kernel<<<dim3(DD / 256, BB), blk, 0, stream>>>(tstep, sm0);
    gemm_small<2><<<dim3(4 * DD / 256, BB), blk, 0, stream>>>(sm0, F(2), F(3), sm1, 4 * DD, DD);
    gemm_small<0><<<dim3(DD / 256, BB),     blk, 0, stream>>>(sm1, F(4), F(5), sm2, DD, 4 * DD);
    ln_kernel<<<BB, blk, 0, stream>>>(sm2, nullptr, nullptr, 1, F(6), F(7), te);

    // ---- input embedding ----------------------------------------------
    gemm_bf16_wmma<1><<<gD, blk, 0, stream>>>(x,  F(8),  F(9),  ta, MM, DD, DD);   // tanh
    gemm_bf16_wmma<0><<<gD, blk, 0, stream>>>(ta, F(10), F(11), tb, MM, DD, DD);
    ln_kernel<<<MM, blk, 0, stream>>>(tb, nullptr, te, SS, F(12), F(13), xe);      // + te bcast

    // ---- transformer layers --------------------------------------------
    for (int l = 0; l < LL; ++l) {
        const int p = 14 + 11 * l;
        gemm_bf16_wmma<0><<<g3D, blk, 0, stream>>>(xe, F(p+0), F(p+1), qkv, MM, 3*DD, DD);
        attn_kernel<<<MM, blk, 0, stream>>>(qkv, ta);
        gemm_bf16_wmma<0><<<gD, blk, 0, stream>>>(ta, F(p+2), nullptr, tb, MM, DD, DD);
        ln_kernel<<<MM, blk, 0, stream>>>(xe, tb, nullptr, 1, F(p+3), F(p+4), xe); // ln1
        gemm_bf16_wmma<0><<<gD, blk, 0, stream>>>(xe, F(p+7), F(p+8),  ta, MM, DD, DD);
        gemm_bf16_wmma<0><<<gD, blk, 0, stream>>>(ta, F(p+9), F(p+10), tb, MM, DD, DD);
        ln_kernel<<<MM, blk, 0, stream>>>(xe, tb, nullptr, 1, F(p+5), F(p+6), xe); // ln2
    }

    // ---- output head ----------------------------------------------------
    gemm_bf16_wmma<1><<<gD, blk, 0, stream>>>(xe, F(102), F(103), ta, MM, DD, DD); // tanh
    gemm_bf16_wmma<0><<<gD, blk, 0, stream>>>(ta, F(104), F(105), tb, MM, DD, DD);
    ln_kernel<<<MM, blk, 0, stream>>>(tb, nullptr, nullptr, 1, F(106), F(107), (float*)d_out);
}
